// CRF_85512798863707
// MI455X (gfx1250) — matrix operand order repaired and checked by reading, not compiled, not run on hardware
//
#include <hip/hip_runtime.h>
#include <hip/hip_bf16.h>
#include <cstdint>

#define LBL   512
#define BATCH 64
#define SEQ   512

typedef _Float16 v16h __attribute__((ext_vector_type(16)));
typedef _Float16 v8h  __attribute__((ext_vector_type(8)));
typedef float    v8f  __attribute__((ext_vector_type(8)));

// ---------------- prep: column max of T, pre-swizzled f16 exp(T - mt) ----------------

__global__ void crf_colmax(const float* __restrict__ T, float* __restrict__ mt) {
  int j = blockIdx.x * blockDim.x + threadIdx.x;
  if (j < LBL) {
    float m = -3.4e38f;
    for (int i = 0; i < LBL; ++i) m = fmaxf(m, T[i * LBL + j]);
    mt[j] = m;
  }
}

// Layout: tile = nb*16 + kb (nb: 16-wide N block, kb: 32-deep K block).
// Within a tile, lane l holds 16 contiguous f16: column n = nb*16 + (l&15),
// K = kb*32 + ((l>>4)<<4) + e  -- contiguous k order this code also uses for A,
// so the forward kernel loads each B fragment as one contiguous 32B read.
__global__ void crf_expT(const float* __restrict__ T, const float* __restrict__ mt,
                         _Float16* __restrict__ expTsw) {
  int idx    = blockIdx.x * blockDim.x + threadIdx.x;  // 0 .. 262143
  int within = idx & 511;
  int tile   = idx >> 9;
  int l  = within >> 4;
  int e  = within & 15;
  int kb = tile & 15;
  int nb = tile >> 4;
  int K  = kb * 32 + ((l >> 4) << 4) + e;
  int n  = nb * 16 + (l & 15);
  expTsw[idx] = (_Float16)__expf(T[K * LBL + n] - mt[n]);
}

// ---------------- forward scan (persistent, WMMA f16 -> f32) ----------------

__device__ inline void grid_barrier(unsigned* cnt, int nwg) {
  __threadfence();
  __syncthreads();
  if (threadIdx.x == 0) {
    unsigned arrive = atomicAdd(cnt, 1u);
    unsigned target = (arrive / (unsigned)nwg + 1u) * (unsigned)nwg;
    while (atomicAdd(cnt, 0u) < target) {
      __builtin_amdgcn_s_sleep(2);
    }
  }
  __syncthreads();
  __threadfence();
}

// 8 persistent WGs x 256 threads = 8 waves/WG -> 2 waves per SIMD32, so the
// matrix pipe interleaves 4 independent WMMA accumulator chains per SIMD
// (f16 WMMA->WMMA RAW needs ~5 slots; 1 wave with 2 chains would stall).
__launch_bounds__(256, 1)
__global__ void crf_forward(const float* __restrict__ x,
                            const float* __restrict__ startT,
                            const float* __restrict__ endT,
                            const _Float16* __restrict__ expTsw,
                            const float* __restrict__ mt,
                            float* __restrict__ logits,
                            unsigned* __restrict__ barCnt) {
  __shared__ float    aSh[16 * LBL];      // 32 KB: prev-step logits (this batch block)
  __shared__ _Float16 eaSh[16 * LBL];     // 16 KB: exp-shifted f16 (WMMA A source)
  __shared__ float    mtSh[LBL];          //  2 KB: col max of T
  __shared__ float    maSh[16];
  __shared__ float    pmax[16][16];

  const int tid  = threadIdx.x;
  const int wg   = blockIdx.x;         // 8 WGs persistent
  const int mb   = wg >> 1;            // batch block: rows mb*16 .. +15
  const int ng   = wg & 1;             // 256-wide label group
  const int wave = tid >> 5;           // 0..7
  const int lane = tid & 31;
  const int nb0  = ng * 16 + wave * 2; // two 16-wide N tiles per wave

  for (int i = tid; i < LBL; i += 256) mtSh[i] = mt[i];

  // t = 0 : l0 = x[:,0] + start_transitions  (this WG's 16x256 slice)
  for (int i = tid; i < 16 * 256; i += 256) {
    int r = i >> 8;
    int c = (i & 255) + ng * 256;
    size_t off = (size_t)(mb * 16 + r) * SEQ * LBL + c;
    logits[off] = x[off] + startT[c];
  }
  grid_barrier(barCnt, 8);

  for (int t = 1; t < SEQ; ++t) {
    // prefetch this step's x slice (consumed in the epilogue, much later)
    {
      int r = tid >> 4;
      int c = ng * 256 + (tid & 15) * 16;
      __builtin_prefetch(&x[(size_t)(mb * 16 + r) * SEQ * LBL + (size_t)t * LBL + c], 0, 3);
    }
    // stage prev logits rows (L2-resident) into LDS
    for (int i = tid; i < 16 * LBL; i += 256) {
      int r = i >> 9;
      int c = i & 511;
      aSh[i] = logits[(size_t)(mb * 16 + r) * SEQ * LBL + (size_t)(t - 1) * LBL + c];
    }
    __syncthreads();
    // row maxes: 16 lanes per row, then 16-thread final reduce
    {
      int r = tid >> 4, sl = tid & 15;
      float m = -3.4e38f;
      for (int c = sl; c < LBL; c += 16) m = fmaxf(m, aSh[r * LBL + c]);
      pmax[r][sl] = m;
    }
    __syncthreads();
    if (tid < 16) {
      float m = pmax[tid][0];
#pragma unroll
      for (int k = 1; k < 16; ++k) m = fmaxf(m, pmax[tid][k]);
      maSh[tid] = m;
    }
    __syncthreads();
    // exp-shift to f16: values in (0,1], safe for f16 A operand
    for (int i = tid; i < 16 * LBL; i += 256) {
      int r = i >> 9;
      eaSh[i] = (_Float16)__expf(aSh[i] - maSh[r]);
    }
    __syncthreads();

    // D(16x16) += A(16x32 f16) x B(32x16 f16), K = 512 -> 16 WMMAs per tile
    v8f acc0 = {};
    v8f acc1 = {};
    const int rA = lane & 15;
#pragma unroll 4
    for (int kb = 0; kb < 16; ++kb) {
      // A fragment from LDS: lane holds row rA, 16 contiguous K (16h .. 16h+15), matching B
      int kBase = kb * 32 + ((lane >> 4) << 4);
      v8h lo = *(const v8h*)&eaSh[rA * LBL + kBase];
      v8h hi = *(const v8h*)&eaSh[rA * LBL + kBase + 8];
      v16h afrag = __builtin_shufflevector(lo, hi, 0, 1, 2, 3, 4, 5, 6, 7,
                                           8, 9, 10, 11, 12, 13, 14, 15);
      // B fragments: contiguous 32B per lane from the pre-swizzled exp(T) table
      const v16h bf0 = *(const v16h*)&expTsw[(size_t)((nb0 * 16 + kb) * 512) + lane * 16];
      const v16h bf1 = *(const v16h*)&expTsw[(size_t)(((nb0 + 1) * 16 + kb) * 512) + lane * 16];
      acc0 = __builtin_amdgcn_wmma_f32_16x16x32_f16(false, afrag, false, bf0,
                                                    (short)0, acc0, false, false);
      acc1 = __builtin_amdgcn_wmma_f32_16x16x32_f16(false, afrag, false, bf1,
                                                    (short)0, acc1, false, false);
    }

    // epilogue: ma + mt + log(sum) + x[:,t]  (+ end_transitions at final step)
    const bool last = (t == SEQ - 1);
#pragma unroll
    for (int v = 0; v < 8; ++v) {
      int M = v + ((lane >> 4) << 3);                 // C/D layout: VGPR v, lane half
      size_t base = (size_t)(mb * 16 + M) * SEQ * LBL + (size_t)t * LBL;
      int j0 = nb0 * 16 + (lane & 15);
      float val0 = maSh[M] + mtSh[j0] + __logf(acc0[v]) + x[base + j0];
      if (last) val0 += endT[j0];
      logits[base + j0] = val0;
      int j1 = (nb0 + 1) * 16 + (lane & 15);
      float val1 = maSh[M] + mtSh[j1] + __logf(acc1[v]) + x[base + j1];
      if (last) val1 += endT[j1];
      logits[base + j1] = val1;
    }
    grid_barrier(barCnt, 8);
  }
}

// ---------------- backward argmax walk: one wave32 per batch row ----------------

__global__ void crf_backward(const float* __restrict__ T,
                             const float* __restrict__ logits,
                             int* __restrict__ seq) {
  const int wave = threadIdx.x >> 5;
  const int lane = threadIdx.x & 31;
  const int b = blockIdx.x * (blockDim.x >> 5) + wave;
  if (b >= BATCH) return;
  const float* lg = logits + (size_t)b * SEQ * LBL;

  // last_tag = argmax over final row (first-index tie-break like jnp.argmax)
  float best = -3.4e38f;
  int bi = LBL;
  for (int i = lane; i < LBL; i += 32) {
    float v = lg[(size_t)(SEQ - 1) * LBL + i];
    if (v > best) { best = v; bi = i; }
  }
#pragma unroll
  for (int off = 16; off > 0; off >>= 1) {
    float ov = __shfl_xor(best, off, 32);
    int   oi = __shfl_xor(bi, off, 32);
    if (ov > best || (ov == best && oi < bi)) { best = ov; bi = oi; }
  }
  if (lane == 0) seq[b * SEQ + (SEQ - 1)] = bi;

  for (int t = SEQ - 2; t >= 0; --t) {
    const int ntag = bi;                 // identical in all lanes after butterfly
    const float* row = lg + (size_t)t * LBL;
    best = -3.4e38f;
    bi = LBL;
    for (int i = lane; i < LBL; i += 32) {
      float v = row[i] + T[i * LBL + ntag];   // Tt[next_tag][i] == T[i][next_tag]
      if (v > best) { best = v; bi = i; }
    }
#pragma unroll
    for (int off = 16; off > 0; off >>= 1) {
      float ov = __shfl_xor(best, off, 32);
      int   oi = __shfl_xor(bi, off, 32);
      if (ov > best || (ov == best && oi < bi)) { best = ov; bi = oi; }
    }
    if (lane == 0) seq[b * SEQ + t] = bi;
  }
}

// ---------------- host side ----------------

extern "C" void kernel_launch(void* const* d_in, const int* in_sizes, int n_in,
                              void* d_out, int out_size, void* d_ws, size_t ws_size,
                              hipStream_t stream) {
  (void)in_sizes; (void)n_in; (void)out_size; (void)ws_size;
  const float* x      = (const float*)d_in[0];
  const float* T      = (const float*)d_in[1];
  const float* startT = (const float*)d_in[2];
  const float* endT   = (const float*)d_in[3];
  // d_in[4] = y : unused for MAP decode

  char* ws = (char*)d_ws;
  unsigned* barCnt = (unsigned*)ws;                              // 256 B (zeroed below)
  _Float16* expTsw = (_Float16*)(ws + 256);                      // 512 KB
  float*    mtBuf  = (float*)(ws + 256 + 512 * 1024);            // 2 KB
  float*    logits = (float*)(ws + 256 + 512 * 1024 + 2048);     // 64 MB [B][S][L]

  hipMemsetAsync(d_ws, 0, 256, stream);  // graph-capturable barrier reset
  crf_colmax<<<2, 256, 0, stream>>>(T, mtBuf);
  crf_expT<<<(LBL * LBL) / 256, 256, 0, stream>>>(T, mtBuf, expTsw);
  crf_forward<<<8, 256, 0, stream>>>(x, startT, endT, expTsw, mtBuf, logits, barCnt);
  crf_backward<<<8, 256, 0, stream>>>(T, logits, (int*)d_out);
}
